// AblationModel_70978629533898
// MI455X (gfx1250) — matrix operand order repaired and checked by reading, not compiled, not run on hardware
//
#include <hip/hip_runtime.h>
#include <hip/hip_bf16.h>
#include <cstdint>

typedef _Float16 v16h __attribute__((ext_vector_type(16)));
typedef _Float16 v8h  __attribute__((ext_vector_type(8)));
typedef float    v8f  __attribute__((ext_vector_type(8)));

#define BB 16
#define NN 8192

// ---------------------------------------------------------------------------
// Weight convert: f32 (K x N) row-major -> f16 transposed (N x Kpad), zero pad
// ---------------------------------------------------------------------------
__global__ void cvt_w(const float* __restrict__ W, int K, int N, int Kp, int Np,
                      _Float16* __restrict__ dst) {
  int i = blockIdx.x * 256 + threadIdx.x;
  if (i >= Kp * Np) return;
  int n = i / Kp, k = i % Kp;
  float v = (k < K && n < N) ? W[(long long)k * N + n] : 0.0f;
  dst[i] = (_Float16)v;
}

// ---------------------------------------------------------------------------
// Extract xyz (first 3 of 6 channels) into tightly packed (B,N,3)
// ---------------------------------------------------------------------------
__global__ void extract_xyz(const float* __restrict__ x, float* __restrict__ xyz) {
  int i = blockIdx.x * 256 + threadIdx.x;
  if (i >= BB * NN) return;
  xyz[i * 3 + 0] = x[(long long)i * 6 + 0];
  xyz[i * 3 + 1] = x[(long long)i * 6 + 1];
  xyz[i * 3 + 2] = x[(long long)i * 6 + 2];
}

// ---------------------------------------------------------------------------
// Farthest point sampling: one block per batch
// ---------------------------------------------------------------------------
__global__ __launch_bounds__(1024) void fps_kernel(const float* __restrict__ xyz,
                                                   int N, int npoint,
                                                   int* __restrict__ out) {
  int b = blockIdx.x, tid = threadIdx.x;
  const float* p = xyz + (long long)b * N * 3;
  float dist[8];
#pragma unroll
  for (int u = 0; u < 8; ++u) dist[u] = 1e10f;
  __shared__ float rv[1024];
  __shared__ int   ri[1024];
  __shared__ float cen[3];
  int far = 0;
  for (int it = 0; it < npoint; ++it) {
    if (tid == 0) {
      out[b * npoint + it] = far;
      cen[0] = p[far * 3 + 0]; cen[1] = p[far * 3 + 1]; cen[2] = p[far * 3 + 2];
    }
    __syncthreads();
    float cx = cen[0], cy = cen[1], cz = cen[2];
    float bv = -1.0f; int bi = 0;
#pragma unroll
    for (int u = 0; u < 8; ++u) {
      int j = tid + u * 1024;
      if (j < N) {
        float dx = p[j * 3] - cx, dy = p[j * 3 + 1] - cy, dz = p[j * 3 + 2] - cz;
        float d2 = dx * dx + dy * dy + dz * dz;
        float dm = fminf(dist[u], d2);
        dist[u] = dm;
        if (dm > bv) { bv = dm; bi = j; }
      }
    }
    rv[tid] = bv; ri[tid] = bi;
    __syncthreads();
    for (int s = 512; s > 0; s >>= 1) {
      if (tid < s) {
        float ov = rv[tid + s]; int oi = ri[tid + s];
        if (ov > rv[tid] || (ov == rv[tid] && oi < ri[tid])) { rv[tid] = ov; ri[tid] = oi; }
      }
      __syncthreads();
    }
    far = ri[0];
  }
}

// ---------------------------------------------------------------------------
// Gather xyz rows by index: dst[b,s,:] = src[b, idx[b,s], :]
// ---------------------------------------------------------------------------
__global__ void gather_xyz(const float* __restrict__ src, const int* __restrict__ idx,
                           float* __restrict__ dst, int Nsrc, int S) {
  int r = blockIdx.x * 256 + threadIdx.x;
  if (r >= BB * S) return;
  int b = r / S;
  int j = idx[r];
  const float* p = src + (long long)(b * Nsrc + j) * 3;
  dst[r * 3 + 0] = p[0]; dst[r * 3 + 1] = p[1]; dst[r * 3 + 2] = p[2];
}

// ---------------------------------------------------------------------------
// Ball query: one wave per query point; keep lowest nsample indices in radius
// (equivalent to reference sort-then-truncate), pad with first hit.
// ---------------------------------------------------------------------------
__global__ void ball_query(const float* __restrict__ xyz, const float* __restrict__ cent,
                           int N, int S, float r2, int nsample, int* __restrict__ gidx) {
  int w = (blockIdx.x * blockDim.x + threadIdx.x) >> 5;
  int lane = threadIdx.x & 31;
  if (w >= BB * S) return;
  int b = w / S;
  const float* c = cent + (long long)w * 3;
  float cx = c[0], cy = c[1], cz = c[2];
  const float* p = xyz + (long long)b * N * 3;
  int* outp = gidx + (long long)w * nsample;
  int cnt = 0, first = 0;
  bool gotfirst = false;
  for (int j0 = 0; j0 < N && cnt < nsample; j0 += 32) {
    int j = j0 + lane;
    float dx = p[j * 3] - cx, dy = p[j * 3 + 1] - cy, dz = p[j * 3 + 2] - cz;
    bool in = (dx * dx + dy * dy + dz * dz) <= r2;
    unsigned m = (unsigned)__ballot(in);
    if (m) {
      if (!gotfirst) { first = j0 + (__ffs(m) - 1); gotfirst = true; }
      if (in) {
        int pos = cnt + __popc(m & ((1u << lane) - 1u));
        if (pos < nsample) outp[pos] = j;
      }
      cnt += __popc(m);
    }
  }
  for (int t = cnt + lane; t < nsample; t += 32) outp[t] = first;
}

// ---------------------------------------------------------------------------
// SA gather kernels -> padded f16 activation rows
// ---------------------------------------------------------------------------
__global__ void sa1_gather(const float* __restrict__ x, const float* __restrict__ l1xyz,
                           const int* __restrict__ gidx, _Float16* __restrict__ A) {
  int r = blockIdx.x * 256 + threadIdx.x;           // B*512*32
  if (r >= BB * 512 * 32) return;
  int b = r >> 14, rem = r & 16383, s = rem >> 5;
  int j = gidx[r];
  const float* cen = l1xyz + (long long)(b * 512 + s) * 3;
  const float* pj = x + (long long)(b * NN + j) * 6;
  _Float16* row = A + (long long)r * 32;
  row[0] = (_Float16)(pj[0] - cen[0]);
  row[1] = (_Float16)(pj[1] - cen[1]);
  row[2] = (_Float16)(pj[2] - cen[2]);
  for (int c = 0; c < 6; ++c) row[3 + c] = (_Float16)pj[c];
  for (int c = 9; c < 32; ++c) row[c] = (_Float16)0.0f;
}

__global__ void sa2_gather(const float* __restrict__ l1xyz, const float* __restrict__ l2xyz,
                           const _Float16* __restrict__ l1pts, const int* __restrict__ gidx,
                           _Float16* __restrict__ A) {
  int r = blockIdx.x * 256 + threadIdx.x;           // B*128*64
  if (r >= BB * 128 * 64) return;
  int b = r >> 13, rem = r & 8191, s = rem >> 6;
  int j = gidx[r];
  const float* cen = l2xyz + (long long)(b * 128 + s) * 3;
  const float* pj = l1xyz + (long long)(b * 512 + j) * 3;
  const _Float16* f = l1pts + (long long)(b * 512 + j) * 128;
  _Float16* row = A + (long long)r * 160;
  row[0] = (_Float16)(pj[0] - cen[0]);
  row[1] = (_Float16)(pj[1] - cen[1]);
  row[2] = (_Float16)(pj[2] - cen[2]);
  for (int c = 0; c < 128; ++c) row[3 + c] = f[c];
  for (int c = 131; c < 160; ++c) row[c] = (_Float16)0.0f;
}

__global__ void sa3_gather(const float* __restrict__ l2xyz, const _Float16* __restrict__ l2pts,
                           _Float16* __restrict__ A) {
  int r = blockIdx.x * 256 + threadIdx.x;           // B*128
  if (r >= BB * 128) return;
  const float* pj = l2xyz + (long long)r * 3;
  const _Float16* f = l2pts + (long long)r * 256;
  _Float16* row = A + (long long)r * 288;
  row[0] = (_Float16)pj[0]; row[1] = (_Float16)pj[1]; row[2] = (_Float16)pj[2];
  for (int c = 0; c < 256; ++c) row[3 + c] = f[c];
  for (int c = 259; c < 288; ++c) row[c] = (_Float16)0.0f;
}

__global__ void fp3_gather(const _Float16* __restrict__ l2pts, const _Float16* __restrict__ l3pts,
                           _Float16* __restrict__ A) {
  int r = blockIdx.x * 256 + threadIdx.x;           // B*128
  if (r >= BB * 128) return;
  int b = r >> 7;
  _Float16* row = A + (long long)r * 1280;
  const _Float16* f = l2pts + (long long)r * 256;
  for (int c = 0; c < 256; ++c) row[c] = f[c];
  const _Float16* g = l3pts + (long long)b * 1024;
  for (int c = 0; c < 1024; ++c) row[256 + c] = g[c];
}

// ---------------------------------------------------------------------------
// 3-NN interpolation gathers (fp2, fp1)
// ---------------------------------------------------------------------------
__global__ void fp2_gather(const float* __restrict__ l1xyz, const float* __restrict__ l2xyz,
                           const _Float16* __restrict__ l1pts, const _Float16* __restrict__ fp3out,
                           _Float16* __restrict__ A) {
  int r = blockIdx.x * 256 + threadIdx.x;           // B*512
  if (r >= BB * 512) return;
  int b = r >> 9;
  float px = l1xyz[r * 3], py = l1xyz[r * 3 + 1], pz = l1xyz[r * 3 + 2];
  float d0 = 1e30f, d1 = 1e30f, d2 = 1e30f; int i0 = 0, i1 = 0, i2 = 0;
  for (int j = 0; j < 128; ++j) {
    const float* q = l2xyz + (long long)(b * 128 + j) * 3;
    float dx = px - q[0], dy = py - q[1], dz = pz - q[2];
    float d = dx * dx + dy * dy + dz * dz;
    if (d < d0)      { d2 = d1; i2 = i1; d1 = d0; i1 = i0; d0 = d; i0 = j; }
    else if (d < d1) { d2 = d1; i2 = i1; d1 = d;  i1 = j; }
    else if (d < d2) { d2 = d;  i2 = j; }
  }
  float w0 = 1.0f / (d0 + 1e-8f), w1 = 1.0f / (d1 + 1e-8f), w2 = 1.0f / (d2 + 1e-8f);
  float ws = w0 + w1 + w2; w0 /= ws; w1 /= ws; w2 /= ws;
  _Float16* row = A + (long long)r * 384;
  const _Float16* p1 = l1pts + (long long)r * 128;
  for (int c = 0; c < 128; ++c) row[c] = p1[c];
  const _Float16* q0 = fp3out + (long long)(b * 128 + i0) * 256;
  const _Float16* q1 = fp3out + (long long)(b * 128 + i1) * 256;
  const _Float16* q2 = fp3out + (long long)(b * 128 + i2) * 256;
  for (int c = 0; c < 256; ++c)
    row[128 + c] = (_Float16)(w0 * (float)q0[c] + w1 * (float)q1[c] + w2 * (float)q2[c]);
}

__global__ void fp1_gather(const float* __restrict__ x, const float* __restrict__ l0xyz,
                           const float* __restrict__ l1xyz, const _Float16* __restrict__ fp2out,
                           _Float16* __restrict__ A) {
  int r = blockIdx.x * 256 + threadIdx.x;           // B*8192
  if (r >= BB * NN) return;
  int b = r >> 13;
  float px = l0xyz[r * 3], py = l0xyz[r * 3 + 1], pz = l0xyz[r * 3 + 2];
  float d0 = 1e30f, d1 = 1e30f, d2 = 1e30f; int i0 = 0, i1 = 0, i2 = 0;
  for (int j = 0; j < 512; ++j) {
    const float* q = l1xyz + (long long)(b * 512 + j) * 3;
    float dx = px - q[0], dy = py - q[1], dz = pz - q[2];
    float d = dx * dx + dy * dy + dz * dz;
    if (d < d0)      { d2 = d1; i2 = i1; d1 = d0; i1 = i0; d0 = d; i0 = j; }
    else if (d < d1) { d2 = d1; i2 = i1; d1 = d;  i1 = j; }
    else if (d < d2) { d2 = d;  i2 = j; }
  }
  float w0 = 1.0f / (d0 + 1e-8f), w1 = 1.0f / (d1 + 1e-8f), w2 = 1.0f / (d2 + 1e-8f);
  float ws = w0 + w1 + w2; w0 /= ws; w1 /= ws; w2 /= ws;
  _Float16* row = A + (long long)r * 160;
  const float* pj = x + (long long)r * 6;
  for (int c = 0; c < 6; ++c) row[c] = (_Float16)pj[c];
  const _Float16* q0 = fp2out + (long long)(b * 512 + i0) * 128;
  const _Float16* q1 = fp2out + (long long)(b * 512 + i1) * 128;
  const _Float16* q2 = fp2out + (long long)(b * 512 + i2) * 128;
  for (int c = 0; c < 128; ++c)
    row[6 + c] = (_Float16)(w0 * (float)q0[c] + w1 * (float)q1[c] + w2 * (float)q2[c]);
  for (int c = 134; c < 160; ++c) row[c] = (_Float16)0.0f;
}

// ---------------------------------------------------------------------------
// WMMA GEMM + affine-BN + ReLU:  out = relu((A @ W + b) * s + t)
// A: f16 M x Kpad row-major; Wt: f16 Npad x Kpad (transposed); out: f16 M x Npad
// One wave32 per 16x16 output tile, K-loop step 32 via v_wmma_f32_16x16x32_f16.
// ---------------------------------------------------------------------------
__global__ void gemm_bn_act(const _Float16* __restrict__ A, const _Float16* __restrict__ Wt,
                            const float* __restrict__ bias, const float* __restrict__ scale,
                            const float* __restrict__ shift, _Float16* __restrict__ out,
                            int M, int Kpad, int Nact, int Npad) {
  int wave = (blockIdx.x * blockDim.x + threadIdx.x) >> 5;
  int lane = threadIdx.x & 31;
  int tilesN = Npad >> 4;
  long long tiles = (long long)(M >> 4) * tilesN;
  if (wave >= tiles) return;                       // wave-uniform
  int tm = wave / tilesN, tn = wave % tilesN;

  const _Float16* arow = A  + (long long)(tm * 16 + (lane & 15)) * Kpad + ((lane >> 4) * 16);
  const _Float16* bcol = Wt + (long long)(tn * 16 + (lane & 15)) * Kpad + ((lane >> 4) * 16);

  v8f acc = {};
  for (int k0 = 0; k0 < Kpad; k0 += 32) {
    v8h alo = *(const v8h*)(arow + k0);
    v8h ahi = *(const v8h*)(arow + k0 + 8);
    v16h a = __builtin_shufflevector(alo, ahi, 0, 1, 2, 3, 4, 5, 6, 7,
                                     8, 9, 10, 11, 12, 13, 14, 15);
    v16h bfrag = *(const v16h*)(bcol + k0);
    __builtin_prefetch(arow + k0 + 64, 0, 0);      // -> global_prefetch_b8
    acc = __builtin_amdgcn_wmma_f32_16x16x32_f16(false, a, false, bfrag,
                                                 (short)0, acc, false, false);
  }
  int n = (tn << 4) + (lane & 15);
  int mbase = (tm << 4) + ((lane >> 4) << 3);
  float bb = 0.0f, ss = 1.0f, tt = 0.0f;
  if (n < Nact) { bb = bias[n]; ss = scale[n]; tt = shift[n]; }
  _Float16* orow = out + (long long)mbase * Npad + n;
#pragma unroll
  for (int i = 0; i < 8; ++i) {
    float v = (acc[i] + bb) * ss + tt;
    v = fmaxf(v, 0.0f);
    orow[(long long)i * Npad] = (_Float16)v;
  }
}

// ---------------------------------------------------------------------------
// Neighborhood max-pool, staging group tile in LDS via CDNA5 async-to-LDS.
// One block per (b, s) group of NS rows x NP channels.
// ---------------------------------------------------------------------------
template <int NS, int NP>
__global__ void maxpool_async(const _Float16* __restrict__ src, _Float16* __restrict__ dst) {
  __shared__ _Float16 tile[NS * NP];
  int g = blockIdx.x;
  const char* gsrc = (const char*)(src + (long long)g * NS * NP);
  unsigned lbase = (unsigned)(uintptr_t)&tile[0];
  constexpr int BYTES = NS * NP * 2;
  for (int off = threadIdx.x * 16; off < BYTES; off += 256 * 16) {
    unsigned laddr = lbase + (unsigned)off;
    unsigned long long ga = (unsigned long long)(gsrc + off);
    asm volatile("global_load_async_to_lds_b128 %0, %1, off"
                 :: "v"(laddr), "v"(ga) : "memory");
  }
  asm volatile("s_wait_asynccnt 0x0" ::: "memory");
  __syncthreads();
  for (int c = threadIdx.x; c < NP; c += 256) {
    float m = -1e30f;
#pragma unroll 4
    for (int k = 0; k < NS; ++k) m = fmaxf(m, (float)tile[k * NP + c]);
    dst[(long long)g * NP + c] = (_Float16)m;
  }
}

__global__ void maxpool_generic(const _Float16* __restrict__ src, _Float16* __restrict__ dst,
                                int groups, int NS, int NP) {
  long long i = (long long)blockIdx.x * 256 + threadIdx.x;
  if (i >= (long long)groups * NP) return;
  int g = (int)(i / NP), c = (int)(i % NP);
  const _Float16* p = src + (long long)g * NS * NP + c;
  float m = -1e30f;
  for (int k = 0; k < NS; ++k) m = fmaxf(m, (float)p[(long long)k * NP]);
  dst[i] = (_Float16)m;
}

// ---------------------------------------------------------------------------
// Final classifier head (128 -> 2, f32 output) and identity matrices
// ---------------------------------------------------------------------------
__global__ void head2_kernel(const _Float16* __restrict__ feat, const float* __restrict__ W,
                             const float* __restrict__ b, float* __restrict__ out) {
  int r = blockIdx.x * 256 + threadIdx.x;
  if (r >= BB * NN) return;
  const _Float16* f = feat + (long long)r * 128;
  float a0 = b[0], a1 = b[1];
  for (int c = 0; c < 128; ++c) {
    float v = (float)f[c];
    a0 += v * W[c * 2 + 0];
    a1 += v * W[c * 2 + 1];
  }
  out[(long long)r * 2 + 0] = a0;
  out[(long long)r * 2 + 1] = a1;
}

__global__ void eye_kernel(float* __restrict__ out) {
  int i = blockIdx.x * 256 + threadIdx.x;
  if (i >= 2 * BB * 9) return;
  int m = i % 9;
  out[i] = (m == 0 || m == 4 || m == 8) ? 1.0f : 0.0f;
}

// ---------------------------------------------------------------------------
// Host orchestration
// ---------------------------------------------------------------------------
extern "C" void kernel_launch(void* const* d_in, const int* in_sizes, int n_in,
                              void* d_out, int out_size, void* d_ws, size_t ws_size,
                              hipStream_t stream) {
  (void)in_sizes; (void)n_in; (void)out_size; (void)ws_size;
  const float* x = (const float*)d_in[0];

  // Layer tables (insertion order of setup_inputs params):
  // sa1 W:1-3 b:4-6 s:7-9 t:10-12 | sa2:13-24 | sa3:25-36 | fp3:37-44 (2 layers)
  // fp2:45-56 | fp1:57-68 | head1:69-72 | head2 W:73 b:74
  static const int wi[18] = {1,2,3, 13,14,15, 25,26,27, 37,38, 45,46,47, 57,58,59, 69};
  static const int bi[18] = {4,5,6, 16,17,18, 28,29,30, 39,40, 48,49,50, 60,61,62, 70};
  static const int si[18] = {7,8,9, 19,20,21, 31,32,33, 41,42, 51,52,53, 63,64,65, 71};
  static const int ti[18] = {10,11,12, 22,23,24, 34,35,36, 43,44, 54,55,56, 66,67,68, 72};
  static const int Kk[18] = {9,64,64, 131,128,128, 259,256,512, 1280,256, 384,256,256, 134,128,128, 128};
  static const int Nn[18] = {64,64,128, 128,128,256, 256,512,1024, 256,256, 256,256,128, 128,128,128, 128};
  static const int Kp[18] = {32,64,64, 160,128,128, 288,256,512, 1280,256, 384,256,256, 160,128,128, 128};
  static const int Np[18] = {64,64,128, 128,128,256, 256,512,1024, 256,256, 256,256,128, 128,128,128, 128};

  // Workspace carve-out
  char* ws = (char*)d_ws;
  size_t cur = 0;
  auto carve = [&](size_t bytes) -> char* {
    char* p = ws + cur;
    cur = (cur + bytes + 255) & ~((size_t)255);
    return p;
  };
  _Float16* wt[18];
  for (int l = 0; l < 18; ++l) wt[l] = (_Float16*)carve((size_t)Kp[l] * Np[l] * 2);

  float* XYZ0   = (float*)carve((size_t)BB * NN * 3 * 4);
  float* L1XYZ  = (float*)carve((size_t)BB * 512 * 3 * 4);
  float* L2XYZ  = (float*)carve((size_t)BB * 128 * 3 * 4);
  int*   FPS1   = (int*)carve((size_t)BB * 512 * 4);
  int*   FPS2   = (int*)carve((size_t)BB * 128 * 4);
  int*   GIDX1  = (int*)carve((size_t)BB * 512 * 32 * 4);
  int*   GIDX2  = (int*)carve((size_t)BB * 128 * 64 * 4);
  _Float16* L1PTS  = (_Float16*)carve((size_t)BB * 512 * 128 * 2);
  _Float16* L2PTS  = (_Float16*)carve((size_t)BB * 128 * 256 * 2);
  _Float16* L3PTS  = (_Float16*)carve((size_t)BB * 1024 * 2);
  _Float16* FP3OUT = (_Float16*)carve((size_t)BB * 128 * 256 * 2);
  _Float16* FP2OUT = (_Float16*)carve((size_t)BB * 512 * 128 * 2);
  _Float16* ACTA = (_Float16*)carve((size_t)131072 * 160 * 2);  // 42 MB
  _Float16* ACTB = (_Float16*)carve((size_t)262144 * 128 * 2);  // 67 MB
  _Float16* ACTC = (_Float16*)carve((size_t)262144 * 64 * 2);   // 34 MB

  // Weight conversion (18 small launches)
  for (int l = 0; l < 18; ++l) {
    int tot = Kp[l] * Np[l];
    cvt_w<<<(tot + 255) / 256, 256, 0, stream>>>((const float*)d_in[wi[l]],
                                                 Kk[l], Nn[l], Kp[l], Np[l], wt[l]);
  }

  auto gemm = [&](const _Float16* A, int l, _Float16* out, int M) {
    long long waves = (long long)(M / 16) * (Np[l] / 16);
    int blocks = (int)((waves + 7) / 8);
    gemm_bn_act<<<blocks, 256, 0, stream>>>(A, wt[l],
        (const float*)d_in[bi[l]], (const float*)d_in[si[l]], (const float*)d_in[ti[l]],
        out, M, Kp[l], Nn[l], Np[l]);
  };

  // ---- level 0 -> 1 (sa1) ----
  extract_xyz<<<(BB * NN + 255) / 256, 256, 0, stream>>>(x, XYZ0);
  fps_kernel<<<BB, 1024, 0, stream>>>(XYZ0, NN, 512, FPS1);
  gather_xyz<<<(BB * 512 + 255) / 256, 256, 0, stream>>>(XYZ0, FPS1, L1XYZ, NN, 512);
  ball_query<<<(BB * 512 * 32 + 255) / 256, 256, 0, stream>>>(XYZ0, L1XYZ, NN, 512, 0.04f, 32, GIDX1);
  sa1_gather<<<(BB * 512 * 32 + 255) / 256, 256, 0, stream>>>(x, L1XYZ, GIDX1, ACTA);
  gemm(ACTA, 0, ACTB, 262144);
  gemm(ACTB, 1, ACTC, 262144);
  gemm(ACTC, 2, ACTB, 262144);
  maxpool_async<32, 128><<<BB * 512, 256, 0, stream>>>(ACTB, L1PTS);

  // ---- level 1 -> 2 (sa2) ----
  fps_kernel<<<BB, 1024, 0, stream>>>(L1XYZ, 512, 128, FPS2);
  gather_xyz<<<(BB * 128 + 255) / 256, 256, 0, stream>>>(L1XYZ, FPS2, L2XYZ, 512, 128);
  ball_query<<<(BB * 128 * 32 + 255) / 256, 256, 0, stream>>>(L1XYZ, L2XYZ, 512, 128, 0.16f, 64, GIDX2);
  sa2_gather<<<(BB * 128 * 64 + 255) / 256, 256, 0, stream>>>(L1XYZ, L2XYZ, L1PTS, GIDX2, ACTA);
  gemm(ACTA, 3, ACTB, 131072);
  gemm(ACTB, 4, ACTC, 131072);
  gemm(ACTC, 5, ACTB, 131072);
  maxpool_async<64, 256><<<BB * 128, 256, 0, stream>>>(ACTB, L2PTS);

  // ---- level 2 -> 3 (sa3, global) ----
  sa3_gather<<<(BB * 128 + 255) / 256, 256, 0, stream>>>(L2XYZ, L2PTS, ACTA);
  gemm(ACTA, 6, ACTB, 2048);
  gemm(ACTB, 7, ACTC, 2048);
  gemm(ACTC, 8, ACTB, 2048);
  maxpool_generic<<<(BB * 1024 + 255) / 256, 256, 0, stream>>>(ACTB, L3PTS, BB, 128, 1024);

  // ---- fp3 ----
  fp3_gather<<<(BB * 128 + 255) / 256, 256, 0, stream>>>(L2PTS, L3PTS, ACTA);
  gemm(ACTA, 9, ACTB, 2048);
  gemm(ACTB, 10, FP3OUT, 2048);

  // ---- fp2 ----
  fp2_gather<<<(BB * 512 + 255) / 256, 256, 0, stream>>>(L1XYZ, L2XYZ, L1PTS, FP3OUT, ACTA);
  gemm(ACTA, 11, ACTB, 8192);
  gemm(ACTB, 12, ACTC, 8192);
  gemm(ACTC, 13, FP2OUT, 8192);

  // ---- fp1 ----
  fp1_gather<<<(BB * NN + 255) / 256, 256, 0, stream>>>(x, XYZ0, L1XYZ, FP2OUT, ACTA);
  gemm(ACTA, 14, ACTB, 131072);
  gemm(ACTB, 15, ACTC, 131072);
  gemm(ACTC, 16, ACTB, 131072);

  // ---- heads ----
  gemm(ACTB, 17, ACTC, 131072);  // head1 (128->128)
  head2_kernel<<<(BB * NN + 255) / 256, 256, 0, stream>>>(ACTC,
      (const float*)d_in[73], (const float*)d_in[74], (float*)d_out);
  eye_kernel<<<2, 256, 0, stream>>>((float*)d_out + (size_t)BB * NN * 2);
}
